// ConvDU_45346264711758
// MI455X (gfx1250) — matrix operand order repaired and checked by reading, not compiled, not run on hardware
//
#include <hip/hip_runtime.h>

// ---------------------------------------------------------------------------
// ConvDU vertical recurrence on MI455X (gfx1250, wave32, WMMA)
//
// Each row step:  Y[w][co] = relu( sum_{k,ci} A[w+k-4][ci] * Wk[ci][co] + b[co] ) + addend
// computed as bf16 WMMA (16x16x32) with f32 accumulate.
//   A operand : transposed row state  [w][ci]   (bf16, LDS w/ +-4 zero halo)
//   B operand : weights pre-packed in exact per-lane fragment order (global)
//   D         : M = w, N = co  -> writes next transposed state directly
// ---------------------------------------------------------------------------

typedef __attribute__((ext_vector_type(16))) __bf16 v16bf;
typedef __attribute__((ext_vector_type(8)))  float  v8f;

#define NB 8
#define CC 256
#define HH 96
#define WWD 96
#define KW 9

union Frag32 { uint4 u[2]; v16bf v; };   // 32 bytes = 16 bf16 per lane

// ---- pack W (co, ci, k) f32 -> per-lane B-fragment order bf16 -------------
// Wfrag[(k*8+cib)*16 + cot][lane][e] = W[cot*16 + lane%16][cib*32 + (lane/16)*16 + e][k]
__global__ void prep_weights(const float* __restrict__ W, __bf16* __restrict__ Wfrag) {
    int f = blockIdx.x * blockDim.x + threadIdx.x;       // 0 .. 589823
    if (f >= KW * 8 * 16 * 512) return;
    int fragid = f >> 9;
    int e      = f & 511;
    int lane   = e >> 4;
    int el     = e & 15;
    int k   = fragid / (8 * 16);
    int cib = (fragid >> 4) & 7;
    int cot = fragid & 15;
    int co  = cot * 16 + (lane & 15);
    int ci  = cib * 32 + ((lane >> 4) << 4) + el;
    Wfrag[f] = (__bf16)W[(co * CC + ci) * KW + k];
}

// ---- out row 0 = fea row 0 ------------------------------------------------
__global__ void copy_row0(const float* __restrict__ fea, float* __restrict__ out) {
    int idx = blockIdx.x * blockDim.x + threadIdx.x;
    if (idx >= NB * CC * WWD) return;
    int n = idx / (CC * WWD);
    int rem = idx % (CC * WWD);
    int c = rem / WWD, w = rem % WWD;
    size_t off = (((size_t)n * CC + c) * HH + 0) * WWD + w;
    out[off] = fea[off];
}

// ---- encode (n,c,h,w) f32 row h -> transposed bf16 state [n][w][c] --------
__global__ void encode_row(const float* __restrict__ src, int h, __bf16* __restrict__ dst) {
    int n = blockIdx.x;
    for (int p = threadIdx.x; p < CC * WWD; p += blockDim.x) {
        int c = p / WWD, w = p % WWD;
        float v = src[(((size_t)n * CC + c) * HH + h) * WWD + w];
        dst[(size_t)n * WWD * CC + (size_t)w * CC + c] = (__bf16)v;
    }
}

// ---- one recurrence step: conv + bias + relu + addend ---------------------
// grid = 32 blocks (n in 0..7, co-block of 64 in 0..3), 128 threads = 4 waves.
// Each wave owns one 16-wide co tile and 6 w-tiles (6 C accumulators).
__global__ __launch_bounds__(128) void conv_step(
    const __bf16* __restrict__ Ain,    // [n][w][c] bf16 (prev row state, transposed)
    const float*  __restrict__ addend, // full (n,c,h,w) f32 (fea for down, out for up)
    float*        __restrict__ out,    // full (n,c,h,w) f32
    __bf16*       __restrict__ Aout,   // [n][w][c] bf16 (this row, transposed)
    const __bf16* __restrict__ Wfrag,
    const float*  __restrict__ bias,
    int h)
{
    __shared__ __align__(16) __bf16 As[104 * CC];   // rows w=-4..99 (halo), 53,248 B
    __shared__ __align__(16) float  RowS[64 * WWD]; // addend in, then result out, 24,576 B

    const int tid  = threadIdx.x;
    const int nb   = blockIdx.x >> 2;
    const int cb   = blockIdx.x & 3;
    const int wv   = tid >> 5;
    const int lane = tid & 31;
    const int m    = lane & 15;   // A row (w) / C column (co) selector
    const int hi   = lane >> 4;

    // Phase 1: stage transposed prev-state row into LDS (+ zero halo)
    {
        uint4* dl = (uint4*)As;
        const uint4 z = {0u, 0u, 0u, 0u};
        dl[tid]        = z;        // front halo: 4*256 bf16 = 128 uint4
        dl[3200 + tid] = z;        // back  halo: starts at elem 100*256
        const uint4* src = (const uint4*)(Ain + (size_t)nb * WWD * CC);
        for (int q = tid; q < 1536; q += 128)      // 1536 uint4 = 48*256 bf16 (rows w=0..47 only)
            dl[128 + q] = src[q];
    }
    // Phase 2: stage addend rows (co-block x 96) f32, coalesced
    for (int q = tid; q < 1536; q += 128) {        // 64*96 f32 = 1536 uint4
        int r  = q / 24;
        int wq = (q % 24) * 4;
        const float* p = addend + (((size_t)nb * CC + cb * 64 + r) * HH + h) * WWD + wq;
        ((uint4*)RowS)[q] = *(const uint4*)p;
    }
    __syncthreads();

    // Phase 3: GEMM  D[w][co] += A[w+k-4][ci] * Wk[ci][co]
    v8f acc[6] = {};
    const int cotg = cb * 4 + wv;                  // global 16-wide co tile
    for (int k = 0; k < KW; ++k) {
        for (int cib = 0; cib < 8; ++cib) {
            Frag32 bf;
            const uint4* bp = (const uint4*)(Wfrag +
                ((size_t)(((k * 8 + cib) * 16) + cotg) * 512 + lane * 16));
            bf.u[0] = bp[0];
            bf.u[1] = bp[1];
#pragma unroll
            for (int wt = 0; wt < 6; ++wt) {
                Frag32 af;
                int row = wt * 16 + m + k;                 // halo (+4) cancels (k-4)
                int off = row * CC + cib * 32 + hi * 16;
                af.u[0] = *(const uint4*)&As[off];
                af.u[1] = *(const uint4*)&As[off + 8];
                acc[wt] = __builtin_amdgcn_wmma_f32_16x16x32_bf16(
                    false, af.v, false, bf.v, (short)0, acc[wt], false, false);
            }
        }
    }

    // Phase 4: epilogue  v = relu(acc + b) + addend ; emit f32 + transposed bf16
    const int rr = wv * 16 + m;                    // row within 64-wide co block
    const float bval = bias[cb * 64 + rr];
    __bf16* aout_base = Aout + (size_t)nb * WWD * CC + cb * 64 + rr;
#pragma unroll
    for (int wt = 0; wt < 6; ++wt) {
#pragma unroll
        for (int j = 0; j < 8; ++j) {
            int w = wt * 16 + j + hi * 8;          // D: M = j + 8*hi
            float v = acc[wt][j] + bval;
            v = v > 0.0f ? v : 0.0f;
            int idx = rr * WWD + w;                // element uniquely owned by this lane
            v += RowS[idx];
            RowS[idx] = v;                         // reuse stage for output
            aout_base[(size_t)w * CC] = (__bf16)v; // next-step A operand
        }
    }
    __syncthreads();
    // coalesced f32 writeback to (n,c,h,w)
    for (int q = tid; q < 1536; q += 128) {
        int r  = q / 24;
        int wq = (q % 24) * 4;
        float* p = out + (((size_t)nb * CC + cb * 64 + r) * HH + h) * WWD + wq;
        *(uint4*)p = ((uint4*)RowS)[q];
    }
}

// ---------------------------------------------------------------------------
extern "C" void kernel_launch(void* const* d_in, const int* in_sizes, int n_in,
                              void* d_out, int out_size, void* d_ws, size_t ws_size,
                              hipStream_t stream) {
    (void)in_sizes; (void)n_in; (void)out_size; (void)ws_size;
    const float* fea = (const float*)d_in[0];
    const float* W   = (const float*)d_in[1];
    const float* b   = (const float*)d_in[2];
    float* out = (float*)d_out;

    char* ws = (char*)d_ws;
    __bf16* Wfrag = (__bf16*)ws;                            // 1,179,648 B
    __bf16* A0    = (__bf16*)(ws + 1179648);                //   393,216 B
    __bf16* A1    = (__bf16*)(ws + 1179648 + 393216);       //   393,216 B

    prep_weights<<<2304, 256, 0, stream>>>(W, Wfrag);
    copy_row0<<<(NB * CC * WWD + 255) / 256, 256, 0, stream>>>(fea, out);
    encode_row<<<NB, 256, 0, stream>>>(fea, 0, A0);

    __bf16* Ai = A0; __bf16* Ao = A1;
    // Down pass: s[i] = relu(conv(s[i-1]) + b) + fea[i]
    for (int i = 1; i < HH; ++i) {
        conv_step<<<32, 128, 0, stream>>>(Ai, fea, out, Ao, Wfrag, b, i);
        __bf16* t = Ai; Ai = Ao; Ao = t;
    }
    // Up pass: s[i] = relu(conv(s[i+1]) + b) + s_down[i]  (in place on out)
    encode_row<<<NB, 256, 0, stream>>>(out, HH - 1, Ai);
    for (int i = HH - 2; i >= 0; --i) {
        conv_step<<<32, 128, 0, stream>>>(Ai, out, out, Ao, Wfrag, b, i);
        __bf16* t = Ai; Ai = Ao; Ao = t;
    }
}
